// BSquareModelCombined_24326694765040
// MI455X (gfx1250) — matrix operand order repaired and checked by reading, not compiled, not run on hardware
//
#include <hip/hip_runtime.h>
#include <hip/hip_bf16.h>

// ---------------------------------------------------------------------------
// Types matching the CDNA5 WMMA builtin signatures
// ---------------------------------------------------------------------------
typedef __attribute__((ext_vector_type(16))) __bf16 bf16x16;
typedef __attribute__((ext_vector_type(8)))  float  f32x8;

union FragBF16 {
    uint4          q[2];   // two 16-byte chunks (8 bf16 each)
    unsigned int   w[8];
    unsigned short s[16];
    bf16x16        v;
};

// fp32 -> bf16 (RNE-ish) scalar helper for epilogues
static __device__ __forceinline__ unsigned short f2bf(float f) {
    union { float f; unsigned int u; } c; c.f = f;
    unsigned int lsb = (c.u >> 16) & 1u;
    return (unsigned short)((c.u + 0x7FFFu + lsb) >> 16);
}

// pack two fp32 into (bf16(hi)<<16)|bf16(lo) with one v_perm_b32
static __device__ __forceinline__ unsigned int pack2bf(float lo, float hi) {
    unsigned int a = __float_as_uint(lo) + 0x8000u;   // round half up
    unsigned int b = __float_as_uint(hi) + 0x8000u;
    // D.b[1:0] = src1.b[3:2] (lo), D.b[3:2] = src0.b[3:2] (hi)
    return __builtin_amdgcn_perm(b, a, 0x07060302u);
}

// ---------------------------------------------------------------------------
// Problem constants
// ---------------------------------------------------------------------------
#define NUM_CLASSES 32
#define TRI   496          // 32*31/2
#define HID   15872        // 32*TRI
#define IN_DIM 512
#define BATCH 256
#define NOUT  (2*TRI)      // 992

// GEMM tiling
#define BM 128
#define BN 128
#define BK 32
#define LDSK 48            // padded K-stride (ushorts) -> 96B rows, 16B aligned

// ---------------------------------------------------------------------------
// Utility kernels
// ---------------------------------------------------------------------------
__global__ void k_f32_to_bf16(const float* __restrict__ src,
                              unsigned short* __restrict__ dst, int n) {
    int i = blockIdx.x * blockDim.x + threadIdx.x;
    if (i < n) dst[i] = f2bf(src[i]);
}

__global__ void k_zero_f32(float* __restrict__ p, int n) {
    int i = blockIdx.x * blockDim.x + threadIdx.x;
    if (i < n) p[i] = 0.0f;
}

// ---------------------------------------------------------------------------
// Stage one BK x BN fp32 weight tile into LDS, transposed + bf16-converted.
// Each thread handles 2 K-rows x 4 N-cols per pass (2 passes): pairs along K
// are packed with v_perm_b32 and stored as single dwords -> WT[n][k],WT[n][k+1].
// ---------------------------------------------------------------------------
static __device__ __forceinline__ void stage_tile(
    const float* __restrict__ W, unsigned short* __restrict__ T,
    int N, int kbase, int blockN, int tid)
{
    const int kp = tid >> 5;          // 0..7
    const int wn = (tid & 31) * 4;    // 0..124
#pragma unroll
    for (int p = 0; p < 2; ++p) {
        const int k = (kp + p * 8) * 2;   // even: 0..30
        const float4 r0 = *reinterpret_cast<const float4*>(
            W + (size_t)(kbase + k) * N + blockN + wn);
        const float4 r1 = *reinterpret_cast<const float4*>(
            W + (size_t)(kbase + k + 1) * N + blockN + wn);
        *reinterpret_cast<unsigned int*>(T + (wn + 0) * LDSK + k) = pack2bf(r0.x, r1.x);
        *reinterpret_cast<unsigned int*>(T + (wn + 1) * LDSK + k) = pack2bf(r0.y, r1.y);
        *reinterpret_cast<unsigned int*>(T + (wn + 2) * LDSK + k) = pack2bf(r0.z, r1.z);
        *reinterpret_cast<unsigned int*>(T + (wn + 3) * LDSK + k) = pack2bf(r0.w, r1.w);
    }
}

// ---------------------------------------------------------------------------
// Fused GEMM + bias + ReLU:  C[M,N](bf16) = relu(A[M,K](bf16) * W[K,N](f32)+b)
// fp32 weights are the unavoidable HBM traffic; converted to bf16 on the way
// into LDS (double-buffered), math on v_wmma_f32_16x16x32_bf16 (fp32 accum).
// Block: 256 threads = 8 waves, 2x4 (M x N); wave tile 64x32 = 8 accumulators.
// ---------------------------------------------------------------------------
__global__ void __launch_bounds__(256)
k_gemm_bias_relu(const unsigned short* __restrict__ A,  // M x K  bf16
                 const float*          __restrict__ W,  // K x N  f32
                 const float*          __restrict__ bias,
                 unsigned short*       __restrict__ C,  // M x N  bf16
                 int M, int N, int K)
{
    __shared__ unsigned short WT[2][BN * LDSK];  // double-buffered WT[n][k]

    const int tid    = threadIdx.x;
    const int wid    = tid >> 5;
    const int lane   = tid & 31;
    const int half   = lane >> 4;    // K sub-block select (ISA operand layout)
    const int l16    = lane & 15;

    const int blockN = blockIdx.x * BN;
    const int blockM = blockIdx.y * BM;
    const int wave_m = wid >> 2;     // 0..1  -> 64 rows
    const int wave_n = wid & 3;      // 0..3  -> 32 cols

    f32x8 acc[4][2];
#pragma unroll
    for (int i = 0; i < 4; ++i)
#pragma unroll
        for (int j = 0; j < 2; ++j)
            acc[i][j] = (f32x8){0.f,0.f,0.f,0.f,0.f,0.f,0.f,0.f};

    // prologue: stage first tile into buffer 0
    stage_tile(W, &WT[0][0], N, 0, blockN, tid);

    int buf = 0;
    for (int k0 = 0; k0 < K; k0 += BK, buf ^= 1) {
        __syncthreads();
        // stage next tile into the other buffer while we compute this one
        if (k0 + BK < K)
            stage_tile(W, &WT[buf ^ 1][0], N, k0 + BK, blockN, tid);
        // prefetch the tile after that (global_prefetch_b8)
        if (k0 + 2 * BK < K)
            __builtin_prefetch(W + (size_t)(k0 + 2 * BK + (tid >> 5) * 2) * N
                                 + blockN + (tid & 31) * 4, 0, 0);

        // ---- A fragments straight from global bf16 (layout-exact b128s) ----
        FragBF16 a[4];
#pragma unroll
        for (int ti = 0; ti < 4; ++ti) {
            int m = blockM + wave_m * 64 + ti * 16 + l16;
            const uint4* ap = reinterpret_cast<const uint4*>(
                A + (size_t)m * K + k0 + half * 8);
            a[ti].q[0] = ap[0];   // K = k0 + half*8      .. +7
            a[ti].q[1] = ap[2];   // K = k0 + 16 + half*8 .. +7
        }
        // ---- B fragments from transposed LDS (contiguous b128 reads) ----
        FragBF16 b[2];
#pragma unroll
        for (int tj = 0; tj < 2; ++tj) {
            int nl = wave_n * 32 + tj * 16 + l16;
            const uint4* bp = reinterpret_cast<const uint4*>(
                &WT[buf][nl * LDSK + half * 8]);
            b[tj].q[0] = bp[0];
            b[tj].q[1] = bp[2];
        }
        // ---- 8 WMMAs per K-step ----
#pragma unroll
        for (int ti = 0; ti < 4; ++ti)
#pragma unroll
            for (int tj = 0; tj < 2; ++tj)
                acc[ti][tj] = __builtin_amdgcn_wmma_f32_16x16x32_bf16(
                    false, a[ti].v, false, b[tj].v,
                    (short)0, acc[ti][tj], false, false);
    }

    // ---- epilogue: bias + ReLU + bf16 store ----
#pragma unroll
    for (int ti = 0; ti < 4; ++ti) {
#pragma unroll
        for (int tj = 0; tj < 2; ++tj) {
            int n = blockN + wave_n * 32 + tj * 16 + l16;
            float bv = bias[n];
#pragma unroll
            for (int r = 0; r < 8; ++r) {
                int m = blockM + wave_m * 64 + ti * 16 + half * 8 + r;
                float v = acc[ti][tj][r] + bv;
                v = v > 0.f ? v : 0.f;
                C[(size_t)m * N + n] = f2bf(v);
            }
        }
    }
}

// ---------------------------------------------------------------------------
// Final layer + vote scatter:
//   out = h2 @ Wout + bout  (256 x 992); col 2p adds to class i(p), col 2p+1
//   to class j(p) (mask is always true at ratio = 0).
// One wave per 16x16 out tile; atomics into pre-zeroed votes[256,32].
// ---------------------------------------------------------------------------
__global__ void __launch_bounds__(256)
k_out_votes(const unsigned short* __restrict__ H,     // 256 x HID  bf16
            const float*          __restrict__ Wo,    // HID x 992  f32
            const float*          __restrict__ bout,  // 992
            float*                __restrict__ votes) // 256 x 32 (zeroed)
{
    const int tid  = threadIdx.x;
    const int wid  = tid >> 5;
    const int lane = tid & 31;
    const int half = lane >> 4;
    const int l16  = lane & 15;

    const int idx    = blockIdx.x * 8 + wid;  // 0..991 (16 x 62 tiles)
    const int tile_m = idx / 62;
    const int tile_n = idx % 62;

    const int m = tile_m * 16 + l16;
    const int n = tile_n * 16 + l16;

    f32x8 acc = (f32x8){0.f,0.f,0.f,0.f,0.f,0.f,0.f,0.f};

    for (int k0 = 0; k0 < HID; k0 += 32) {
        FragBF16 a, b;
        const uint4* ap = reinterpret_cast<const uint4*>(
            H + (size_t)m * HID + k0 + half * 16);
        a.q[0] = ap[0];
        a.q[1] = ap[1];
        // column gather of Wout column n; pack K pairs with v_perm_b32
#pragma unroll
        for (int e = 0; e < 8; ++e) {
            const size_t kk = (size_t)(k0 + half * 16 + 2 * e) * NOUT + n;
            b.w[e] = pack2bf(Wo[kk], Wo[kk + NOUT]);
        }
        acc = __builtin_amdgcn_wmma_f32_16x16x32_bf16(
            false, a.v, false, b.v, (short)0, acc, false, false);
    }

    // pair -> (i, j) class indices for this column
    const int p = n >> 1;
    const int comp = n & 1;
    int ci = 0, base = 0;
    while (base + (NUM_CLASSES - 1 - ci) <= p) {
        base += (NUM_CLASSES - 1 - ci);
        ++ci;
    }
    const int cj  = ci + 1 + (p - base);
    const int cls = comp ? cj : ci;
    const float bv = bout[n];

#pragma unroll
    for (int r = 0; r < 8; ++r) {
        int mm = tile_m * 16 + half * 8 + r;
        atomicAdd(&votes[mm * NUM_CLASSES + cls], acc[r] + bv);
    }
}

// ---------------------------------------------------------------------------
// Launcher
// ---------------------------------------------------------------------------
extern "C" void kernel_launch(void* const* d_in, const int* in_sizes, int n_in,
                              void* d_out, int out_size, void* d_ws, size_t ws_size,
                              hipStream_t stream) {
    (void)in_sizes; (void)n_in; (void)out_size; (void)ws_size;

    const float* x    = (const float*)d_in[0];  // [256, 512]
    const float* W1   = (const float*)d_in[1];  // [512, 15872]
    const float* b1   = (const float*)d_in[2];
    const float* W2   = (const float*)d_in[3];  // [15872, 15872]
    const float* b2   = (const float*)d_in[4];
    const float* Wout = (const float*)d_in[5];  // [15872, 992]
    const float* bout = (const float*)d_in[6];
    float* votes = (float*)d_out;               // [256, 32]

    // workspace layout (bf16 activations)
    unsigned short* xb = (unsigned short*)d_ws;                       // 256 KB
    unsigned short* h1 = (unsigned short*)((char*)d_ws + (256u * 1024u));
    unsigned short* h2 = (unsigned short*)((char*)d_ws + (256u * 1024u)
                                           + (size_t)BATCH * HID * 2u);

    // 1) x -> bf16
    {
        int n = BATCH * IN_DIM;
        k_f32_to_bf16<<<(n + 255) / 256, 256, 0, stream>>>(x, xb, n);
    }
    // 2) h1 = relu(x @ W1 + b1)
    {
        dim3 grid(HID / BN, BATCH / BM);
        k_gemm_bias_relu<<<grid, 256, 0, stream>>>(xb, W1, b1, h1,
                                                   BATCH, HID, IN_DIM);
    }
    // 3) h2 = relu(h1 @ W2 + b2)   -- the 129 GFLOP / 1 GB-weight layer
    {
        dim3 grid(HID / BN, BATCH / BM);
        k_gemm_bias_relu<<<grid, 256, 0, stream>>>(h1, W2, b2, h2,
                                                   BATCH, HID, HID);
    }
    // 4) votes = 0, then final layer + vote scatter
    {
        int n = BATCH * NUM_CLASSES;
        k_zero_f32<<<(n + 255) / 256, 256, 0, stream>>>(votes, n);
        k_out_votes<<<(16 * 62) / 8, 256, 0, stream>>>(h2, Wout, bout, votes);
    }
}
